// PointNet4Layers_43001212567721
// MI455X (gfx1250) — compile-verified, matrix operand order checked by reading, not run on hardware
//
#include <hip/hip_runtime.h>

#define N_NODES 50000
#define ENC 64
#define NB 16
#define BN_EPS 1e-5f
#define SLOPE 0.01f

typedef __bf16 bf16;
typedef bf16  v16bf __attribute__((ext_vector_type(16)));
typedef float v8f   __attribute__((ext_vector_type(8)));

union BfVec { v16bf v; unsigned short u[16]; };

__device__ __forceinline__ unsigned short f2bf(float f) {
  unsigned u = __float_as_uint(f);
  unsigned r = ((u >> 16) & 1u) + 0x7FFFu;        // round-to-nearest-even
  return (unsigned short)((u + r) >> 16);
}
__device__ __forceinline__ unsigned pack2(float lo, float hi) {
  return (unsigned)f2bf(lo) | ((unsigned)f2bf(hi) << 16);
}

// float atomic-max via signed-max / unsigned-min punning (valid with -inf init)
__device__ __forceinline__ void atomic_max_f32(float* addr, float val) {
  if (!(val < 0.0f)) atomicMax((int*)addr, __float_as_int(val));
  else               atomicMin((unsigned int*)addr, __float_as_uint(val));
}

// ---------------------------------------------------------------------------
// Fused edge MLP:  leaky(A @ W1 + b1) @ W2 + b2  -> scatter-max into agg
// One wave handles a 16-edge x 64-output tile with v_wmma_f32_16x16x32_bf16.
// W1/W2 are pre-swizzled in LDS into per-lane B-fragment order so each
// fragment is 32 contiguous bytes (2x ds_load_b128).
// ---------------------------------------------------------------------------
template<int INCH, int KPAD>
__launch_bounds__(256)
__global__ void edge_layer_kernel(const float* __restrict__ hin,   // [N, INCH]
                                  const float* __restrict__ pos,   // [N, 2]
                                  const int*   __restrict__ src,
                                  const int*   __restrict__ dst,
                                  const float* __restrict__ w1,    // [INCH+2, 64]
                                  const float* __restrict__ b1,
                                  const float* __restrict__ w2,    // [64, 64]
                                  const float* __restrict__ b2,
                                  float* __restrict__ agg,         // [N, 64] (-inf init)
                                  int nE)
{
  constexpr int K1  = KPAD / 32;                  // k-steps for GEMM1
  constexpr int STG = (KPAD > 64 ? KPAD : 64);    // staging also hosts 16x64 mid
  // fragment-major weights: [kt][nt][lane][16] bf16, each fragment contiguous
  __shared__ __align__(32) unsigned short w1f[K1 * 4 * 32 * 16];
  __shared__ __align__(32) unsigned short w2f[2  * 4 * 32 * 16];
  __shared__ __align__(16) unsigned short stg[8][16][STG];

  const int tid  = threadIdx.x;
  const int lane = tid & 31;
  const int wv   = tid >> 5;

  // ---- pre-swizzle weights into B-fragment order ----
  for (int p = tid; p < K1 * 4 * 32 * 16; p += 256) {
    int i  = p & 15;
    int ln = (p >> 4) & 31;
    int nt = (p >> 9) & 3;
    int kt = p >> 11;
    int k = kt * 32 + ((ln < 16) ? 0 : 8) + ((i < 8) ? i : 8 + i);
    int n = nt * 16 + (ln & 15);
    w1f[p] = f2bf((k < INCH + 2) ? w1[k * ENC + n] : 0.0f);
  }
  for (int p = tid; p < 2 * 4 * 32 * 16; p += 256) {
    int i  = p & 15;
    int ln = (p >> 4) & 31;
    int nt = (p >> 9) & 3;
    int kt = p >> 11;
    int k = kt * 32 + ((ln < 16) ? 0 : 8) + ((i < 8) ? i : 8 + i);
    int n = nt * 16 + (ln & 15);
    w2f[p] = f2bf(w2[k * ENC + n]);
  }
  __syncthreads();

  const int tile  = blockIdx.x * 8 + wv;
  const int eBase = tile * 16;
  if (eBase >= nE) return;

  // prefetch next block's edge indices (global_prefetch_b8)
  if (eBase + 128 < nE) {
    __builtin_prefetch(&src[eBase + 128], 0, 1);
    __builtin_prefetch(&dst[eBase + 128], 0, 1);
  }

  // ---- build A tile: 16 edges x KPAD bf16, wave-private LDS region ----
  {
    const int r   = lane & 15;
    const int sub = lane >> 4;                     // 2 lanes per edge row
    int e = eBase + r; if (e >= nE) e = nE - 1;
    const int s = src[e], d = dst[e];
    const float2 ps = *(const float2*)&pos[2 * s];
    const float2 pd = *(const float2*)&pos[2 * d];
    const float dx = ps.x - pd.x, dy = ps.y - pd.y;
    unsigned int* rowp = (unsigned int*)&stg[wv][r][0];   // STG/2 uints

    if (INCH == 4) {                               // KPAD = 32
      if (sub == 0) {                              // cols 0..15
        const float4 h4 = *(const float4*)&hin[(long)s * 4];
        rowp[0] = pack2(h4.x, h4.y);
        rowp[1] = pack2(h4.z, h4.w);
        rowp[2] = pack2(dx, dy);
#pragma unroll
        for (int p = 3; p < 8; ++p) rowp[p] = 0u;
      } else {                                     // cols 16..31
#pragma unroll
        for (int p = 8; p < 16; ++p) rowp[p] = 0u;
      }
    } else {                                       // INCH = 64, KPAD = 96
      if (sub == 0) {                              // cols 0..47: 12x float4
#pragma unroll
        for (int q = 0; q < 12; ++q) {
          const float4 h4 = *(const float4*)&hin[(long)s * 64 + q * 4];
          rowp[q * 2 + 0] = pack2(h4.x, h4.y);
          rowp[q * 2 + 1] = pack2(h4.z, h4.w);
        }
      } else {                                     // cols 48..95
#pragma unroll
        for (int q = 0; q < 4; ++q) {              // cols 48..63 from hin
          const float4 h4 = *(const float4*)&hin[(long)s * 64 + 48 + q * 4];
          rowp[24 + q * 2 + 0] = pack2(h4.x, h4.y);
          rowp[24 + q * 2 + 1] = pack2(h4.z, h4.w);
        }
        rowp[32] = pack2(dx, dy);                  // cols 64,65
#pragma unroll
        for (int p = 33; p < 48; ++p) rowp[p] = 0u; // cols 66..95
      }
    }
  }
  // same-wave LDS store->load is in-order (DScnt); region is wave-private.

  const int row  = lane & 15;
  const int koff = (lane < 16) ? 0 : 8;           // ISA 16-bit A/B lane split

  // A fragments for GEMM1: two contiguous 16B chunks each (ds_load_b128 x2)
  BfVec a1[K1];
#pragma unroll
  for (int kt = 0; kt < K1; ++kt) {
    int kb = kt * 32 + koff;
    __builtin_memcpy(&a1[kt].u[0], &stg[wv][row][kb],      16);  // K=kb..kb+7
    __builtin_memcpy(&a1[kt].u[8], &stg[wv][row][kb + 16], 16);  // K=kb+16..kb+23
  }

  // GEMM1 (4 n-tiles) + bias + LeakyReLU
  v8f acc[4];
#pragma unroll
  for (int nt = 0; nt < 4; ++nt) {
    v8f c = {};
    int n = nt * 16 + row;
#pragma unroll
    for (int kt = 0; kt < K1; ++kt) {
      BfVec b;
      __builtin_memcpy(&b, &w1f[((kt * 4 + nt) * 32 + lane) * 16], 32);
      c = __builtin_amdgcn_wmma_f32_16x16x32_bf16(false, a1[kt].v, false, b.v,
                                                  (short)0, c, false, false);
    }
    float bias = b1[n];
#pragma unroll
    for (int i = 0; i < 8; ++i) {
      float x = c[i] + bias;
      c[i] = (x > 0.0f) ? x : x * SLOPE;
    }
    acc[nt] = c;
  }

  // re-quantize mid (16x64) into the same wave-private staging region
#pragma unroll
  for (int nt = 0; nt < 4; ++nt) {
    int n = nt * 16 + row;
#pragma unroll
    for (int i = 0; i < 8; ++i) {
      int m = (lane < 16) ? i : (8 + i);          // ISA f32 C/D layout
      stg[wv][m][n] = f2bf(acc[nt][i]);
    }
  }

  // A fragments for GEMM2 (K = 64)
  BfVec a2[2];
#pragma unroll
  for (int kt = 0; kt < 2; ++kt) {
    int kb = kt * 32 + koff;
    __builtin_memcpy(&a2[kt].u[0], &stg[wv][row][kb],      16);
    __builtin_memcpy(&a2[kt].u[8], &stg[wv][row][kb + 16], 16);
  }

  int dsts[8];
#pragma unroll
  for (int i = 0; i < 8; ++i) {
    int m = (lane < 16) ? i : (8 + i);
    int e = eBase + m; if (e >= nE) e = nE - 1;
    dsts[i] = dst[e];
  }

  // GEMM2 + bias + scatter-max
#pragma unroll
  for (int nt = 0; nt < 4; ++nt) {
    v8f c = {};
    int n = nt * 16 + row;
#pragma unroll
    for (int kt = 0; kt < 2; ++kt) {
      BfVec b;
      __builtin_memcpy(&b, &w2f[((kt * 4 + nt) * 32 + lane) * 16], 32);
      c = __builtin_amdgcn_wmma_f32_16x16x32_bf16(false, a2[kt].v, false, b.v,
                                                  (short)0, c, false, false);
    }
    float bias = b2[n];
#pragma unroll
    for (int i = 0; i < 8; ++i) {
      int m = (lane < 16) ? i : (8 + i);
      if (eBase + m < nE)
        atomic_max_f32(&agg[(long)dsts[i] * ENC + n], c[i] + bias);
    }
  }
}

// ---------------------------------------------------------------------------
__global__ void fill_neginf(float* p, int n) {
  for (int i = blockIdx.x * blockDim.x + threadIdx.x; i < n;
       i += gridDim.x * blockDim.x)
    p[i] = __int_as_float(0xFF800000);
}

__global__ void zero_f32(float* p, int n) {
  int i = blockIdx.x * blockDim.x + threadIdx.x;
  if (i < n) p[i] = 0.0f;
}

// per-channel sum & sumsq (treat -inf sentinel as 0, per reference)
__global__ void bn_stats_kernel(const float* __restrict__ agg,
                                float* __restrict__ stats, int total) {
  __shared__ float ssum[ENC], ssq[ENC];
  int tid = threadIdx.x;
  if (tid < ENC) { ssum[tid] = 0.0f; ssq[tid] = 0.0f; }
  __syncthreads();
  for (int i = blockIdx.x * blockDim.x + tid; i < total;
       i += gridDim.x * blockDim.x) {
    float v = agg[i];
    if (__float_as_uint(v) == 0xFF800000u) v = 0.0f;
    int c = i & (ENC - 1);
    atomicAdd(&ssum[c], v);
    atomicAdd(&ssq[c], v * v);
  }
  __syncthreads();
  if (tid < ENC) {
    atomicAdd(&stats[tid], ssum[tid]);
    atomicAdd(&stats[ENC + tid], ssq[tid]);
  }
}

// BN (training mode, biased var) + LeakyReLU + fused batch-pool scatter-max
__global__ void bn_apply_kernel(const float* __restrict__ agg,
                                const float* __restrict__ stats,
                                const float* __restrict__ g,
                                const float* __restrict__ b,
                                const int*   __restrict__ batch,
                                float* __restrict__ hout,
                                float* __restrict__ pooled,
                                int layer, int total) {
  int i = blockIdx.x * blockDim.x + threadIdx.x;
  if (i >= total) return;
  int c    = i & (ENC - 1);
  int node = i >> 6;
  float v = agg[i];
  if (__float_as_uint(v) == 0xFF800000u) v = 0.0f;
  float mu  = stats[c] * (1.0f / N_NODES);
  float var = stats[ENC + c] * (1.0f / N_NODES) - mu * mu;
  float y = g[c] * (v - mu) * rsqrtf(var + BN_EPS) + b[c];
  y = (y > 0.0f) ? y : y * SLOPE;
  hout[i] = y;
  atomic_max_f32(&pooled[batch[node] * 4 * ENC + layer * ENC + c], y);
}

__global__ void final_kernel(const float* __restrict__ pooled,
                             const float* __restrict__ lw,   // [256, 2]
                             const float* __restrict__ lb,
                             float* __restrict__ out) {
  int t = threadIdx.x;
  if (t >= NB * 2) return;
  int bi = t >> 1, o = t & 1;
  float acc = lb[o];
  for (int c = 0; c < 4 * ENC; ++c) {
    float p = pooled[bi * 4 * ENC + c];
    if (__float_as_uint(p) == 0xFF800000u) p = 0.0f;
    acc += p * lw[c * 2 + o];
  }
  out[bi * 2 + o] = acc;
}

// ---------------------------------------------------------------------------
extern "C" void kernel_launch(void* const* d_in, const int* in_sizes, int n_in,
                              void* d_out, int out_size, void* d_ws, size_t ws_size,
                              hipStream_t stream) {
  const float* pos   = (const float*)d_in[0];
  const float* x     = (const float*)d_in[1];
  const int*   batch = (const int*)d_in[2];
  const int*   eidx  = (const int*)d_in[4];          // [2, E+N] int32
  const int nE = in_sizes[4] / 2;                    // 850000
  const int* src = eidx;
  const int* dst = eidx + nE;

  const float *cw1[4], *cb1[4], *cw2[4], *cb2[4], *bg[4], *bb[4];
  for (int l = 0; l < 4; ++l) {
    int base = 5 + l * 6;
    cw1[l] = (const float*)d_in[base + 0];
    cb1[l] = (const float*)d_in[base + 1];
    cw2[l] = (const float*)d_in[base + 2];
    cb2[l] = (const float*)d_in[base + 3];
    bg[l]  = (const float*)d_in[base + 4];
    bb[l]  = (const float*)d_in[base + 5];
  }
  const float* lw = (const float*)d_in[29];
  const float* lb = (const float*)d_in[30];
  float* out = (float*)d_out;

  // workspace: agg | hA | hB | stats[128] | pooled[B*256]   (~38.5 MB)
  float* ws = (float*)d_ws;
  const size_t per = (size_t)N_NODES * ENC;
  float* agg    = ws;
  float* hA     = ws + per;
  float* hB     = ws + 2 * per;
  float* stats  = ws + 3 * per;
  float* pooled = ws + 3 * per + 128;

  const int NH = N_NODES * ENC;
  const int tiles       = (nE + 15) / 16;
  const int eBlocks     = (tiles + 7) / 8;
  const int applyBlocks = (NH + 255) / 256;

  fill_neginf<<<8, 256, 0, stream>>>(pooled, NB * 4 * ENC);

  float* hprev = nullptr;
  float* hcur  = hA;
  for (int l = 0; l < 4; ++l) {
    fill_neginf<<<4096, 256, 0, stream>>>(agg, NH);
    if (l == 0)
      edge_layer_kernel<4, 32><<<eBlocks, 256, 0, stream>>>(
          x, pos, src, dst, cw1[0], cb1[0], cw2[0], cb2[0], agg, nE);
    else
      edge_layer_kernel<64, 96><<<eBlocks, 256, 0, stream>>>(
          hprev, pos, src, dst, cw1[l], cb1[l], cw2[l], cb2[l], agg, nE);
    zero_f32<<<1, 128, 0, stream>>>(stats, 128);
    bn_stats_kernel<<<1024, 256, 0, stream>>>(agg, stats, NH);
    bn_apply_kernel<<<applyBlocks, 256, 0, stream>>>(
        agg, stats, bg[l], bb[l], batch, hcur, pooled, l, NH);
    hprev = hcur;
    hcur  = (hcur == hA) ? hB : hA;
  }
  final_kernel<<<1, 64, 0, stream>>>(pooled, lw, lb, out);
}
